// RoPEAttention_87273735455413
// MI455X (gfx1250) — matrix operand order repaired and checked by reading, not compiled, not run on hardware
//
#include <hip/hip_runtime.h>
#include <hip/hip_bf16.h>
#include <math.h>

// ---------------------------------------------------------------------------
// RoPE attention for MI455X (gfx1250), bf16 WMMA pipeline with f32 accumulate.
// ~19 GFLOP vs ~17 MB HBM traffic -> compute bound; everything runs through
// v_wmma_f32_16x16x32_bf16. Cross-lane softmax reductions use v_permlane16
// (VALU); row-sums ride the WMMA pipe via P x ones. K/V tiles are staged into
// double-buffered LDS with GLOBAL_LOAD_ASYNC_TO_LDS_B128 (ASYNCcnt-tracked,
// no VGPR transit), shared by all 8 waves of a head-cooperative block.
// ---------------------------------------------------------------------------

typedef __attribute__((ext_vector_type(16))) __bf16 v16bf;
typedef __attribute__((ext_vector_type(8)))  __bf16 v8bf;
typedef __attribute__((ext_vector_type(8)))  float  v8f;

#define EMBED 256
#define NTOK  4096
#define HEADS 4
#define HDIM  64
#define FEATX 64
// ln(10000) * 4/64
#define ROPE_LF 0.5756462732485115f

static __device__ __forceinline__ v16bf cat8(v8bf lo, v8bf hi) {
  union { v16bf v; v8bf h[2]; } u;
  u.h[0] = lo; u.h[1] = hi;
  return u.v;
}

// v_permlane16_b32 lane tables for xor-style exchange within each 16-lane row
// (nibble i of {SL,SH} = source lane for dst lane i).
template<unsigned SL, unsigned SH>
static __device__ __forceinline__ float permf(float x) {
  int xi = __float_as_int(x);
  int r  = __builtin_amdgcn_permlane16(xi, xi, (int)SL, (int)SH, false, false);
  return __int_as_float(r);
}
#define PERM_X1 permf<0x67452301u, 0xEFCDAB89u>   // xor 1
#define PERM_X2 permf<0x54761032u, 0xDCFE98BAu>   // xor 2
#define PERM_X4 permf<0x32107654u, 0xBA98FEDCu>   // xor 4
#define PERM_X8 permf<0xFEDCBA98u, 0x76543210u>   // xor 8

// async copy: 16B global -> LDS, GVS mode (saddr base + 32-bit voffset),
// LDS destination byte offset in the VDST VGPR. Tracked by ASYNCcnt.
static __device__ __forceinline__ void async_ld16(unsigned lds_byte_off,
                                                  const void* gbase,
                                                  unsigned gbyte_off) {
  asm volatile("global_load_async_to_lds_b128 %0, %1, %2"
               :
               : "v"(lds_byte_off), "v"(gbyte_off),
                 "s"((unsigned long long)(uintptr_t)gbase)
               : "memory");
}
static __device__ __forceinline__ void wait_async0() {
  asm volatile("s_wait_asynccnt 0x0" ::: "memory");
}

// A-fragment (16x32 bf16, MxK): lane L holds row M = L%16,
// K = { b..b+15 } with b = (L/16)*16  (two 16B contiguous chunks)
static __device__ __forceinline__ v16bf load_a_frag(const __bf16* row_ptr, int lane) {
  const int b = (lane >> 4) << 4;
  v8bf lo = *(const v8bf*)(row_ptr + b);
  v8bf hi = *(const v8bf*)(row_ptr + b + 8);
  return cat8(lo, hi);
}

// ---------------------------------------------------------------------------
// dtype conversion helpers
// ---------------------------------------------------------------------------
__global__ void cvt_f32_to_bf16(const float* __restrict__ src,
                                __bf16* __restrict__ dst, int n) {
  int i = blockIdx.x * blockDim.x + threadIdx.x;
  if (i < n) dst[i] = (__bf16)src[i];
}

// W stored [in][out]; write Wt[out][in] bf16 so B-fragments are contiguous.
__global__ void transpose_w_bf16(const float* __restrict__ w,
                                 __bf16* __restrict__ wt) {
  int o = blockIdx.x;   // 0..255 (out)
  int i = threadIdx.x;  // 0..255 (in)
  wt[o * EMBED + i] = (__bf16)w[i * EMBED + o];
}

// ---------------------------------------------------------------------------
// Projection GEMM: out[m][n] = sum_k A[m][k]*Wt[n][k] + bias[n]
// MODE 0: Q proj  -> RoPE + 1/sqrt(hd) scale, store bf16 row-major
// MODE 1: K proj  -> RoPE,                    store bf16 row-major
// MODE 2: V proj  -> store bf16 TRANSPOSED Vt[col][tok]
// MODE 3: O proj  -> store f32 row-major (final output)
// One wave computes one 16x16 output tile; K=256 -> 8 WMMA steps.
// ---------------------------------------------------------------------------
template<int MODE>
__global__ __launch_bounds__(256) void proj_gemm(const __bf16* __restrict__ A,
                                                 const __bf16* __restrict__ Bt,
                                                 const float* __restrict__ bias,
                                                 void* __restrict__ outp) {
  const int lane = threadIdx.x & 31;
  const int wave = threadIdx.x >> 5;
  const int tile = blockIdx.x * 8 + wave;   // 0..4095
  const int mt = tile >> 4;                 // 0..255
  const int nt = tile & 15;                 // 0..15
  const int ln = lane & 15;
  const int hi = lane >> 4;
  const int row_a = mt * 16 + ln;
  const int col_b = nt * 16 + ln;
  const int k0b = hi << 4;                  // B-fragment K offset per half-wave

  v8f c = {};
#pragma unroll
  for (int kk = 0; kk < 8; ++kk) {
    v16bf a = load_a_frag(A + row_a * EMBED + kk * 32, lane);
    v16bf b = *(const v16bf*)(Bt + col_b * EMBED + kk * 32 + k0b);
    c = __builtin_amdgcn_wmma_f32_16x16x32_bf16(false, a, false, b,
                                                (short)0, c, false, false);
  }

  const float bv = bias[col_b];

#pragma unroll
  for (int r = 0; r < 8; ++r) {
    const int row = mt * 16 + r + 8 * hi;   // token index
    float x = c[r] + bv;
    if (MODE == 0 || MODE == 1) {
      // axial RoPE: pairs (2i,2i+1) live in adjacent lanes of the C layout;
      // fetch the partner with a single v_permlane16 (xor 1).
      float p = PERM_X1(x);
      const int d = col_b & (HDIM - 1);
      const int i = d >> 1;                 // 0..31
      const float f = __expf(-ROPE_LF * (float)(i & 15));
      const float pos = (i < 16) ? (float)(row & (FEATX - 1))
                                 : (float)(row >> 6);
      float sn, cs;
      __sincosf(pos * f, &sn, &cs);
      x = ((lane & 1) == 0) ? (x * cs - p * sn)   // even column: a*cos - b*sin
                            : (p * sn + x * cs);  // odd  column: a*sin + b*cos
      if (MODE == 0) x *= 0.125f;                 // fold in 1/sqrt(64)
    }
    if (MODE == 3) {
      ((float*)outp)[row * EMBED + col_b] = x;
    } else if (MODE == 2) {
      ((__bf16*)outp)[col_b * NTOK + row] = (__bf16)x;  // transposed V
    } else {
      ((__bf16*)outp)[row * EMBED + col_b] = (__bf16)x;
    }
  }
}

// ---------------------------------------------------------------------------
// Flash attention, head-cooperative blocks:
//   block = (head, 128 query rows); 8 waves x 16 rows each.
//   Per 32-key block: K tile (32x64) and V tile (64x32) staged into
//   double-buffered LDS via async-to-LDS loads (one b128 per thread),
//   s_wait_asynccnt + one __syncthreads per iteration, next tile prefetched
//   into the alternate buffer while the current one is consumed.
//   Per iteration: 4 QK^T WMMA, row-max via v_permlane16, P->LDS->A-fragment
//   (s_wait_dscnt), 1 row-sum WMMA (P x ones), 4 PV WMMA.
// ---------------------------------------------------------------------------
__global__ __launch_bounds__(256) void flash_attn(const __bf16* __restrict__ Q,
                                                  const __bf16* __restrict__ K,
                                                  const __bf16* __restrict__ Vt,
                                                  __bf16* __restrict__ O) {
  __shared__ __bf16 ldsK[2][32][64];   // [buf][key][hdim]      8 KB
  __shared__ __bf16 ldsV[2][64][32];   // [buf][hdim(col)][key] 8 KB
  __shared__ __bf16 pbuf[8][16][40];   // per-wave P relayout   10 KB

  const int tid  = threadIdx.x;
  const int lane = tid & 31;
  const int wave = tid >> 5;
  const int head = blockIdx.x >> 5;     // 0..3
  const int qblk = blockIdx.x & 31;     // 0..31
  const int qt   = qblk * 8 + wave;     // query tile 0..255
  const int ln   = lane & 15;
  const int hi   = lane >> 4;
  const int k0b  = hi << 4;

  const __bf16* kbase = K + head * HDIM;          // [NTOK][EMBED] row-major
  const __bf16* vbase = Vt + head * HDIM * NTOK;  // [HDIM][NTOK]

  // cooperative staging indices: 256 threads x 16B = 4KB per tile
  const int krow = tid >> 3, kc8 = (tid & 7) << 3;   // K: 32 rows x 64 cols
  const int vrow = tid >> 2, vc8 = (tid & 3) << 3;   // V: 64 rows x 32 cols
  unsigned ldsKoff[2], ldsVoff[2];
  ldsKoff[0] = (unsigned)(uintptr_t)&ldsK[0][krow][kc8];
  ldsKoff[1] = (unsigned)(uintptr_t)&ldsK[1][krow][kc8];
  ldsVoff[0] = (unsigned)(uintptr_t)&ldsV[0][vrow][vc8];
  ldsVoff[1] = (unsigned)(uintptr_t)&ldsV[1][vrow][vc8];
  const unsigned kgoff = (unsigned)((krow * EMBED + kc8) * sizeof(__bf16));
  const unsigned vgoff = (unsigned)((vrow * NTOK + vc8) * sizeof(__bf16));

  // Q fragments for this wave's 16 rows (hd=64 -> two 16x32 A fragments)
  const __bf16* qrow = Q + (qt * 16 + ln) * EMBED + head * HDIM;
  const v16bf a0 = load_a_frag(qrow, lane);
  const v16bf a1 = load_a_frag(qrow + 32, lane);

  v16bf ones;
#pragma unroll
  for (int i = 0; i < 16; ++i) ones[i] = (__bf16)1.0f;

  v8f acc[4] = {v8f{}, v8f{}, v8f{}, v8f{}};
  float m[8], l[8];
#pragma unroll
  for (int r = 0; r < 8; ++r) { m[r] = -1e30f; l[r] = 0.0f; }

  // stage tile 0 (async, no VGPR transit)
  async_ld16(ldsKoff[0], kbase, kgoff);
  async_ld16(ldsVoff[0], vbase, vgoff);

  __bf16* pt = &pbuf[wave][0][0];

  for (int jb = 0; jb < NTOK / 32; ++jb) {
    const int buf = jb & 1;
    wait_async0();     // this wave's async stores into LDS have landed
    __syncthreads();   // tile jb visible block-wide; alt buffer free
    if (jb + 1 < NTOK / 32) {
      const unsigned j2b = (unsigned)((jb + 1) * 32 * sizeof(__bf16));
      async_ld16(ldsKoff[buf ^ 1], kbase, kgoff + j2b * EMBED / 2 * 2);
      async_ld16(ldsVoff[buf ^ 1], vbase, vgoff + j2b);
    }

    // K^T B-fragments from LDS (lane ln = key column, contiguous 32B)
    v16bf b00 = *(const v16bf*)&ldsK[buf][ln][k0b];
    v16bf b01 = *(const v16bf*)&ldsK[buf][ln][32 + k0b];
    v16bf b10 = *(const v16bf*)&ldsK[buf][16 + ln][k0b];
    v16bf b11 = *(const v16bf*)&ldsK[buf][16 + ln][32 + k0b];

    v8f s0 = {}, s1 = {};
    s0 = __builtin_amdgcn_wmma_f32_16x16x32_bf16(false, a0, false, b00, (short)0, s0, false, false);
    s0 = __builtin_amdgcn_wmma_f32_16x16x32_bf16(false, a1, false, b01, (short)0, s0, false, false);
    s1 = __builtin_amdgcn_wmma_f32_16x16x32_bf16(false, a0, false, b10, (short)0, s1, false, false);
    s1 = __builtin_amdgcn_wmma_f32_16x16x32_bf16(false, a1, false, b11, (short)0, s1, false, false);

    float alpha[8];
#pragma unroll
    for (int r = 0; r < 8; ++r) {
      // row max across the 16 lanes of this half-wave (pure VALU)
      float mx = fmaxf(s0[r], s1[r]);
      mx = fmaxf(mx, PERM_X8(mx));
      mx = fmaxf(mx, PERM_X4(mx));
      mx = fmaxf(mx, PERM_X2(mx));
      mx = fmaxf(mx, PERM_X1(mx));
      const float mn = fmaxf(m[r], mx);
      alpha[r] = __expf(m[r] - mn);
      m[r] = mn;
      // stash P (C layout) into wave-private LDS for A-layout reload
      pt[(r + 8 * hi) * 40 + ln]      = (__bf16)__expf(s0[r] - mn);
      pt[(r + 8 * hi) * 40 + 16 + ln] = (__bf16)__expf(s1[r] - mn);
    }

#pragma unroll
    for (int cb = 0; cb < 4; ++cb)
#pragma unroll
      for (int r = 0; r < 8; ++r) acc[cb][r] *= alpha[r];

    // wave-private LDS exchange: DS ops are in-order per wave; fence compiler
    asm volatile("s_wait_dscnt 0x0" ::: "memory");

    // reload P as a 16x32 A fragment
    const __bf16* prow = pt + ln * 40;
    const int b = hi << 4;
    v16bf pA = cat8(*(const v8bf*)(prow + b), *(const v8bf*)(prow + b + 8));

    // row-sum on the matrix pipe: ssum = P x ones -> every lane holds its
    // rows' sums (consistent with the bf16 P the PV WMMAs consume)
    v8f ssum = {};
    ssum = __builtin_amdgcn_wmma_f32_16x16x32_bf16(false, pA, false, ones,
                                                   (short)0, ssum, false, false);
#pragma unroll
    for (int r = 0; r < 8; ++r) l[r] = l[r] * alpha[r] + ssum[r];

    // PV: V columns are LDS rows -> contiguous 32B B-fragment loads
#pragma unroll
    for (int cb = 0; cb < 4; ++cb) {
      v16bf vb = *(const v16bf*)&ldsV[buf][cb * 16 + ln][k0b];
      acc[cb] = __builtin_amdgcn_wmma_f32_16x16x32_bf16(false, pA, false, vb,
                                                        (short)0, acc[cb], false, false);
    }
  }

  // normalize and store bf16 (input to the output projection)
  float rl[8];
#pragma unroll
  for (int r = 0; r < 8; ++r) rl[r] = 1.0f / l[r];
#pragma unroll
  for (int r = 0; r < 8; ++r) {
    const int row = qt * 16 + r + 8 * hi;
    __bf16* o = O + row * EMBED + head * HDIM + ln;
    o[0]  = (__bf16)(acc[0][r] * rl[r]);
    o[16] = (__bf16)(acc[1][r] * rl[r]);
    o[32] = (__bf16)(acc[2][r] * rl[r]);
    o[48] = (__bf16)(acc[3][r] * rl[r]);
  }
}

// ---------------------------------------------------------------------------
extern "C" void kernel_launch(void* const* d_in, const int* in_sizes, int n_in,
                              void* d_out, int out_size, void* d_ws, size_t ws_size,
                              hipStream_t stream) {
  const float* q  = (const float*)d_in[0];
  const float* k  = (const float*)d_in[1];
  const float* v  = (const float*)d_in[2];
  const float* wq = (const float*)d_in[3];
  const float* bq = (const float*)d_in[4];
  const float* wk = (const float*)d_in[5];
  const float* bk = (const float*)d_in[6];
  const float* wv = (const float*)d_in[7];
  const float* bv = (const float*)d_in[8];
  const float* wo = (const float*)d_in[9];
  const float* bo = (const float*)d_in[10];
  (void)in_sizes; (void)n_in; (void)out_size; (void)ws_size;

  char* ws = (char*)d_ws;
  size_t off = 0;
  auto wsalloc = [&](size_t bytes) {
    char* p = ws + off;
    off += (bytes + 255) & ~(size_t)255;
    return p;
  };
  const size_t actB = (size_t)NTOK * EMBED * sizeof(__bf16);  // 2 MB
  const size_t wB   = (size_t)EMBED * EMBED * sizeof(__bf16); // 128 KB
  __bf16* qb  = (__bf16*)wsalloc(actB);
  __bf16* kb  = (__bf16*)wsalloc(actB);
  __bf16* vb  = (__bf16*)wsalloc(actB);
  __bf16* wqT = (__bf16*)wsalloc(wB);
  __bf16* wkT = (__bf16*)wsalloc(wB);
  __bf16* wvT = (__bf16*)wsalloc(wB);
  __bf16* woT = (__bf16*)wsalloc(wB);
  __bf16* Qr  = (__bf16*)wsalloc(actB);   // roped+scaled Q, bf16 row-major
  __bf16* Kr  = (__bf16*)wsalloc(actB);   // roped K, bf16 row-major
  __bf16* Vtr = (__bf16*)wsalloc(actB);   // V transposed: [256][4096]
  __bf16* Ob  = (__bf16*)wsalloc(actB);   // attention output, bf16 row-major

  const int nAct = NTOK * EMBED;
  cvt_f32_to_bf16<<<(nAct + 255) / 256, 256, 0, stream>>>(q, qb, nAct);
  cvt_f32_to_bf16<<<(nAct + 255) / 256, 256, 0, stream>>>(k, kb, nAct);
  cvt_f32_to_bf16<<<(nAct + 255) / 256, 256, 0, stream>>>(v, vb, nAct);
  transpose_w_bf16<<<EMBED, EMBED, 0, stream>>>(wq, wqT);
  transpose_w_bf16<<<EMBED, EMBED, 0, stream>>>(wk, wkT);
  transpose_w_bf16<<<EMBED, EMBED, 0, stream>>>(wv, wvT);
  transpose_w_bf16<<<EMBED, EMBED, 0, stream>>>(wo, woT);

  // 4096 wave-tiles / 8 waves per block = 512 blocks
  proj_gemm<0><<<512, 256, 0, stream>>>(qb, wqT, bq, (void*)Qr);
  proj_gemm<1><<<512, 256, 0, stream>>>(kb, wkT, bk, (void*)Kr);
  proj_gemm<2><<<512, 256, 0, stream>>>(vb, wvT, bv, (void*)Vtr);

  // 4 heads x 32 query blocks (128 rows) = 128 blocks of 8 waves
  flash_attn<<<128, 256, 0, stream>>>(Qr, Kr, Vtr, Ob);

  proj_gemm<3><<<512, 256, 0, stream>>>(Ob, woT, bo, d_out);
}
